// MultiHeadAttention_77360950935695
// MI455X (gfx1250) — matrix operand order repaired and checked by reading, not compiled, not run on hardware
//
#include <hip/hip_runtime.h>
#include <hip/hip_bf16.h>

// ---------------------------------------------------------------------------
// MI455X (gfx1250) multi-head attention: bf16 WMMA pipeline
//   1) Q/K/V projections: fp32 GEMM -> bf16 (v_wmma_f32_16x16x32_bf16)
//   2) flash-attention (online softmax) per (b,h), bf16 in/out
//   3) output projection: bf16 GEMM -> fp32 + bias
// gfx1250 paths: v_wmma_f32_16x16x32_bf16, global_load_async_to_lds_b128,
// s_wait_asynccnt, global_prefetch_b8, ds_swizzle reductions.
// ---------------------------------------------------------------------------

typedef __bf16 bf16_t;
typedef __attribute__((ext_vector_type(16))) __bf16 v16bf;
typedef __attribute__((ext_vector_type(8)))  __bf16 v8bf;
typedef __attribute__((ext_vector_type(8)))  float  v8f;

#define LOG2E 1.4426950408889634f
// group-of-32 ds_swizzle, xor-mask butterfly (and_mask=0x1f, or=0)
#define SWZ_XOR(x, m) __int_as_float(__builtin_amdgcn_ds_swizzle(__float_as_int(x), ((m) << 10) | 31))

#if defined(__has_builtin)
#if __has_builtin(__builtin_amdgcn_global_load_async_to_lds_b128) && \
    __has_builtin(__builtin_amdgcn_s_wait_asynccnt)
#define HAVE_ASYNC_LDS 1
#endif
#endif

// async-LDS builtin takes int4-vector pointers in AS1 (global) / AS3 (LDS)
typedef int v4i __attribute__((vector_size(16)));
typedef __attribute__((address_space(1))) v4i v4i_g;
typedef __attribute__((address_space(3))) v4i v4i_l;

__device__ __forceinline__ v4i_g* glob_v4(const void* p) {
  return (v4i_g*)(unsigned long long)p;
}
// generic->LDS: CDNA5 maps LDS generic addresses by truncation to low 32 bits
__device__ __forceinline__ v4i_l* lds_v4(const void* p) {
  return (v4i_l*)(unsigned int)(unsigned long long)p;
}

// ---------------------------------------------------------------------------
// GEMM: Out[M,N] = A[M,K] @ W[K,N] + bias[N]
//   A: fp32 or bf16 (template), W/bias fp32, Out: bf16 or fp32 (template)
//   block = 256 threads (8 waves); block tile 128(M) x 64(N); wave tile 32x32
// ---------------------------------------------------------------------------
template <bool A_BF16, bool OUT_BF16>
__global__ __launch_bounds__(256) void gemm_bias_wmma(
    const void* __restrict__ Av, const float* __restrict__ W,
    const float* __restrict__ bias, void* __restrict__ Outv,
    int M, int N, int K) {
  const int tid  = threadIdx.x;
  const int lane = tid & 31;
  const int wave = tid >> 5;
  const int wm   = wave & 3;   // 4 wave rows   -> 128 M
  const int wn   = wave >> 2;  // 2 wave cols   -> 64  N
  const int lr   = lane & 15;  // row/col within 16-tile
  const int kg   = lane >> 4;  // k-group select (contiguous order: K = kg*16..+15)

  const int m0 = blockIdx.y * 128;
  const int n0 = blockIdx.x * 64;

  __shared__ alignas(16) bf16_t sA[128 * 32];  // [row][k]      row-major
  __shared__ alignas(16) bf16_t sB[64 * 32];   // [n][k]        transposed

  v8f acc[2][2] = {};

  for (int kk = 0; kk < K; kk += 32) {
    // ---- stage A tile: 128x32, 16 elems / thread -------------------------
    {
      const int row  = tid >> 1;
      const int half = tid & 1;
      if constexpr (A_BF16) {
        const bf16_t* A   = (const bf16_t*)Av;
        const bf16_t* src = A + (size_t)(m0 + row) * K + kk + half * 16;
        bf16_t*       dst = sA + row * 32 + half * 16;
#if HAVE_ASYNC_LDS
        // straight byte-copy: use the async LDS DMA path (bypasses VGPRs)
        __builtin_amdgcn_global_load_async_to_lds_b128(glob_v4(src),     lds_v4(dst),     0, 0);
        __builtin_amdgcn_global_load_async_to_lds_b128(glob_v4(src + 8), lds_v4(dst + 8), 0, 0);
#else
        const v8bf* s8 = (const v8bf*)src;
        v8bf x0 = s8[0], x1 = s8[1];
        v8bf* d8 = (v8bf*)dst;
        d8[0] = x0; d8[1] = x1;
#endif
      } else {
        const float* A = (const float*)Av;
        const float4* src = (const float4*)(A + (size_t)(m0 + row) * K + kk + half * 16);
        float4 f0 = src[0], f1 = src[1], f2 = src[2], f3 = src[3];
        bf16_t* dst = sA + row * 32 + half * 16;
        dst[0]  = (bf16_t)f0.x; dst[1]  = (bf16_t)f0.y; dst[2]  = (bf16_t)f0.z; dst[3]  = (bf16_t)f0.w;
        dst[4]  = (bf16_t)f1.x; dst[5]  = (bf16_t)f1.y; dst[6]  = (bf16_t)f1.z; dst[7]  = (bf16_t)f1.w;
        dst[8]  = (bf16_t)f2.x; dst[9]  = (bf16_t)f2.y; dst[10] = (bf16_t)f2.z; dst[11] = (bf16_t)f2.w;
        dst[12] = (bf16_t)f3.x; dst[13] = (bf16_t)f3.y; dst[14] = (bf16_t)f3.z; dst[15] = (bf16_t)f3.w;
      }
      if (kk + 32 < K) {  // prefetch next A tile line into cache
        const size_t eoff = (size_t)(m0 + row) * K + kk + 32 + half * 16;
        __builtin_prefetch((const char*)Av + eoff * (A_BF16 ? 2 : 4), 0, 1);
      }
    }
    // ---- stage W tile transposed: 32(k)x64(n) -> sB[n][k], 8 / thread ----
    {
      const int k    = tid >> 3;
      const int nseg = tid & 7;
      const float4* src = (const float4*)(W + (size_t)(kk + k) * N + n0 + nseg * 8);
      float4 f0 = src[0], f1 = src[1];
      const int nb = nseg * 8;
      sB[(nb + 0) * 32 + k] = (bf16_t)f0.x; sB[(nb + 1) * 32 + k] = (bf16_t)f0.y;
      sB[(nb + 2) * 32 + k] = (bf16_t)f0.z; sB[(nb + 3) * 32 + k] = (bf16_t)f0.w;
      sB[(nb + 4) * 32 + k] = (bf16_t)f1.x; sB[(nb + 5) * 32 + k] = (bf16_t)f1.y;
      sB[(nb + 6) * 32 + k] = (bf16_t)f1.z; sB[(nb + 7) * 32 + k] = (bf16_t)f1.w;
      if (kk + 32 < K)  // prefetch next W tile line
        __builtin_prefetch(W + (size_t)(kk + 32 + k) * N + n0 + nseg * 8, 0, 1);
    }
#if HAVE_ASYNC_LDS
    if constexpr (A_BF16) __builtin_amdgcn_s_wait_asynccnt(0);
#endif
    __syncthreads();

    // ---- fragments -------------------------------------------------------
    v16bf afrag[2], bfrag[2];
#pragma unroll
    for (int mt = 0; mt < 2; ++mt) {
      const v8bf* p = (const v8bf*)(sA + (wm * 32 + mt * 16 + lr) * 32 + kg * 16);
      v8bf lo = p[0], hi = p[1];  // K = kg*16..+15 contiguous
#pragma unroll
      for (int j = 0; j < 8; ++j) { afrag[mt][j] = lo[j]; afrag[mt][8 + j] = hi[j]; }
    }
#pragma unroll
    for (int nt = 0; nt < 2; ++nt) {
      const v16bf* p = (const v16bf*)(sB + (wn * 32 + nt * 16 + lr) * 32 + kg * 16);
      bfrag[nt] = p[0];           // lane = column, 16 contiguous K values
    }
#pragma unroll
    for (int mt = 0; mt < 2; ++mt)
#pragma unroll
      for (int nt = 0; nt < 2; ++nt)
        acc[mt][nt] = __builtin_amdgcn_wmma_f32_16x16x32_bf16(
            false, afrag[mt], false, bfrag[nt], (short)0, acc[mt][nt], false, false);
    __syncthreads();
  }

  // ---- epilogue: add bias, store -----------------------------------------
#pragma unroll
  for (int mt = 0; mt < 2; ++mt) {
    const int row = m0 + wm * 32 + mt * 16 + kg * 8;
#pragma unroll
    for (int nt = 0; nt < 2; ++nt) {
      const int col = n0 + wn * 32 + nt * 16 + lr;
      const float bv = bias[col];
#pragma unroll
      for (int r = 0; r < 8; ++r) {
        const float v = acc[mt][nt][r] + bv;
        if constexpr (OUT_BF16)
          ((bf16_t*)Outv)[(size_t)(row + r) * N + col] = (bf16_t)v;
        else
          ((float*)Outv)[(size_t)(row + r) * N + col] = v;
      }
    }
  }
}

// ---------------------------------------------------------------------------
// Flash attention: per (b,h), online softmax over Sk in 32-key chunks.
// block = 128 threads (4 waves); each wave owns 16 query rows.
// Q/K/V/Z in [B,S,H*64] layout, bf16.
// ---------------------------------------------------------------------------
__global__ __launch_bounds__(128) void flash_attn_wmma(
    const bf16_t* __restrict__ Q, const bf16_t* __restrict__ Kp,
    const bf16_t* __restrict__ Vp, bf16_t* __restrict__ Z) {
  constexpr int S = 2048, D = 1024, HD = 64;
  const int qt = blockIdx.x, h = blockIdx.y, b = blockIdx.z;
  const int tid = threadIdx.x, lane = tid & 31, wave = tid >> 5;
  const int lr = lane & 15, kg = lane >> 4;
  const int q0 = qt * 64 + wave * 16;

  __shared__ alignas(16) bf16_t sK[32 * 64];      // [key][d]   row-major
  __shared__ alignas(16) bf16_t sV[64 * 32];      // [d][key]   transposed
  __shared__ alignas(16) bf16_t sP[4][16 * 32];   // per-wave P transpose buf

  const size_t base = ((size_t)b * S) * D + (size_t)h * HD;

  // Q fragments (two 16x32 d-halves) stay in registers for the whole kernel
  v16bf qfrag[2];
#pragma unroll
  for (int dh = 0; dh < 2; ++dh) {
    const v8bf* p = (const v8bf*)(Q + base + (size_t)(q0 + lr) * D + dh * 32 + kg * 16);
    v8bf lo = p[0], hi = p[1];
#pragma unroll
    for (int j = 0; j < 8; ++j) { qfrag[dh][j] = lo[j]; qfrag[dh][8 + j] = hi[j]; }
  }

  v8f z[4] = {};
  float mrow[8], lrow[8];
#pragma unroll
  for (int r = 0; r < 8; ++r) { mrow[r] = -1e30f; lrow[r] = 0.f; }

  for (int kc = 0; kc < S; kc += 32) {
    __syncthreads();  // protect previous chunk's sK/sV from overwrite
    // ---- stage K (async DMA, row-major) and V (transposed) chunks --------
    {
      const int krow = tid >> 2, seg = tid & 3;
      const bf16_t* ksrc = Kp + base + (size_t)(kc + krow) * D + seg * 16;
      bf16_t*       kdst = sK + krow * 64 + seg * 16;
#if HAVE_ASYNC_LDS
      __builtin_amdgcn_global_load_async_to_lds_b128(glob_v4(ksrc),     lds_v4(kdst),     0, 0);
      __builtin_amdgcn_global_load_async_to_lds_b128(glob_v4(ksrc + 8), lds_v4(kdst + 8), 0, 0);
#else
      const v8bf* ks = (const v8bf*)ksrc;
      v8bf k0 = ks[0], k1 = ks[1];
      v8bf* kd = (v8bf*)kdst;
      kd[0] = k0; kd[1] = k1;
#endif
      // V needs a transpose -> must pass through VGPRs (overlaps the K DMA)
      const v8bf* vs = (const v8bf*)(Vp + base + (size_t)(kc + krow) * D + seg * 16);
      v8bf v0 = vs[0], v1 = vs[1];
#pragma unroll
      for (int j = 0; j < 8; ++j) {
        sV[(seg * 16 + j) * 32 + krow]     = v0[j];
        sV[(seg * 16 + 8 + j) * 32 + krow] = v1[j];
      }
      if (kc + 32 < S) {  // prefetch next 32-key chunk
        __builtin_prefetch(Kp + base + (size_t)(kc + 32 + krow) * D + seg * 16, 0, 1);
        __builtin_prefetch(Vp + base + (size_t)(kc + 32 + krow) * D + seg * 16, 0, 1);
      }
    }
#if HAVE_ASYNC_LDS
    __builtin_amdgcn_s_wait_asynccnt(0);
#endif
    __syncthreads();

    // ---- scores: two 16x16 key tiles, contraction over d=64 (2 wmma each)
    v8f s0 = {}, s1 = {};
#pragma unroll
    for (int dh = 0; dh < 2; ++dh) {
      const v16bf* k0p = (const v16bf*)(sK + lr * 64        + dh * 32 + kg * 16);
      const v16bf* k1p = (const v16bf*)(sK + (16 + lr) * 64 + dh * 32 + kg * 16);
      s0 = __builtin_amdgcn_wmma_f32_16x16x32_bf16(false, qfrag[dh], false, *k0p, (short)0, s0, false, false);
      s1 = __builtin_amdgcn_wmma_f32_16x16x32_bf16(false, qfrag[dh], false, *k1p, (short)0, s1, false, false);
    }

    // ---- online softmax (row = r + 8*kg; reduce across 16-lane half) -----
    v8f p0, p1;
    float corr[8];
#pragma unroll
    for (int r = 0; r < 8; ++r) {
      const float a0 = s0[r] * 0.125f;   // 1/sqrt(64)
      const float a1 = s1[r] * 0.125f;
      float mx = fmaxf(a0, a1);
      mx = fmaxf(mx, SWZ_XOR(mx, 1));
      mx = fmaxf(mx, SWZ_XOR(mx, 2));
      mx = fmaxf(mx, SWZ_XOR(mx, 4));
      mx = fmaxf(mx, SWZ_XOR(mx, 8));
      const float mnew = fmaxf(mrow[r], mx);
      const float e0 = exp2f((a0 - mnew) * LOG2E);
      const float e1 = exp2f((a1 - mnew) * LOG2E);
      float rs = e0 + e1;
      rs += SWZ_XOR(rs, 1);
      rs += SWZ_XOR(rs, 2);
      rs += SWZ_XOR(rs, 4);
      rs += SWZ_XOR(rs, 8);
      const float c = exp2f((mrow[r] - mnew) * LOG2E);
      lrow[r] = lrow[r] * c + rs;
      mrow[r] = mnew;
      corr[r] = c;
      p0[r] = e0; p1[r] = e1;
    }
#pragma unroll
    for (int dt = 0; dt < 4; ++dt)
#pragma unroll
      for (int r = 0; r < 8; ++r) z[dt][r] *= corr[r];

    // ---- P: C-layout -> LDS -> A-layout (same-wave DS ops are in-order) --
    {
      bf16_t* pw = &sP[wave][0];
#pragma unroll
      for (int r = 0; r < 8; ++r) {
        const int m = r + kg * 8;
        pw[m * 32 + lr]      = (bf16_t)p0[r];
        pw[m * 32 + 16 + lr] = (bf16_t)p1[r];
      }
    }
    v16bf pfrag;
    {
      const v8bf* pp = (const v8bf*)(&sP[wave][0] + lr * 32 + kg * 16);
      v8bf lo = pp[0], hi = pp[1];
#pragma unroll
      for (int j = 0; j < 8; ++j) { pfrag[j] = lo[j]; pfrag[8 + j] = hi[j]; }
    }

    // ---- PV: P(16x32) @ V(32x64) -> 4 wmma over d tiles -------------------
#pragma unroll
    for (int dt = 0; dt < 4; ++dt) {
      const v16bf* vp = (const v16bf*)(sV + (dt * 16 + lr) * 32 + kg * 16);
      z[dt] = __builtin_amdgcn_wmma_f32_16x16x32_bf16(false, pfrag, false, *vp, (short)0, z[dt], false, false);
    }
  }

  // ---- normalize and store Z in [B,S,H*64] layout --------------------------
#pragma unroll
  for (int dt = 0; dt < 4; ++dt) {
#pragma unroll
    for (int r = 0; r < 8; ++r) {
      const int m = r + kg * 8;
      const float v = z[dt][r] / lrow[r];
      Z[base + (size_t)(q0 + m) * D + dt * 16 + lr] = (bf16_t)v;
    }
  }
}

// ---------------------------------------------------------------------------
// Host-side pipeline
// ---------------------------------------------------------------------------
extern "C" void kernel_launch(void* const* d_in, const int* in_sizes, int n_in,
                              void* d_out, int out_size, void* d_ws, size_t ws_size,
                              hipStream_t stream) {
  constexpr int B = 4, S = 2048, DM = 1024;
  constexpr size_t TOK = (size_t)B * S;        // 8192 rows
  constexpr size_t MAT = TOK * DM;             // 8 Mi elements

  const float* query = (const float*)d_in[0];
  const float* key   = (const float*)d_in[1];
  const float* value = (const float*)d_in[2];
  const float* Wq = (const float*)d_in[3]; const float* bq = (const float*)d_in[4];
  const float* Wk = (const float*)d_in[5]; const float* bk = (const float*)d_in[6];
  const float* Wv = (const float*)d_in[7]; const float* bv = (const float*)d_in[8];
  const float* Wo = (const float*)d_in[9]; const float* bo = (const float*)d_in[10];
  float* out = (float*)d_out;

  // workspace: Qp | Kp | Vp | Z, all bf16 [B*S, 1024] (4 x 16.8 MB = 67 MB)
  bf16_t* Qp = (bf16_t*)d_ws;
  bf16_t* Kp = Qp + MAT;
  bf16_t* Vp = Kp + MAT;
  bf16_t* Zb = Vp + MAT;

  dim3 gGemm(DM / 64, TOK / 128);  // (16, 64)
  gemm_bias_wmma<false, true><<<gGemm, 256, 0, stream>>>(query, Wq, bq, Qp, (int)TOK, DM, DM);
  gemm_bias_wmma<false, true><<<gGemm, 256, 0, stream>>>(key,   Wk, bk, Kp, (int)TOK, DM, DM);
  gemm_bias_wmma<false, true><<<gGemm, 256, 0, stream>>>(value, Wv, bv, Vp, (int)TOK, DM, DM);

  flash_attn_wmma<<<dim3(S / 64, 16, B), 128, 0, stream>>>(Qp, Kp, Vp, Zb);

  gemm_bias_wmma<true, false><<<gGemm, 256, 0, stream>>>(Zb, Wo, bo, out, (int)TOK, DM, DM);
}
